// LSTM_ONE_51007031607820
// MI455X (gfx1250) — compile-verified, matrix operand order checked by reading, not run on hardware
//
#include <hip/hip_runtime.h>
#include <hip/hip_bf16.h>

typedef __attribute__((ext_vector_type(16))) __bf16         v16bf;
typedef __attribute__((ext_vector_type(16))) unsigned short v16u;
typedef __attribute__((ext_vector_type(8)))  unsigned short v8u;
typedef __attribute__((ext_vector_type(8)))  float          v8f;
typedef __attribute__((ext_vector_type(4)))  float          v4f;

#define TSTEPS 512
#define BATCH  64
#define IDIM   512
#define HDIM   512

// ---------- helpers ----------

__device__ __forceinline__ unsigned short f2bf(float f) {
  unsigned u = __float_as_uint(f);
  u += 0x7FFFu + ((u >> 16) & 1u);          // round-to-nearest-even
  return (unsigned short)(u >> 16);
}

// K index held by (element i, lane) for 16-bit WMMA A/B operands (ISA 7.12.2):
// lanes 0-15 hold K {0..7, 16..23}, lanes 16-31 hold K {8..15, 24..31}.
__device__ __forceinline__ int kOf(int i, int lane) {
  return ((lane & 16) ? 8 : 0) + ((i < 8) ? i : i + 8);
}

__device__ __forceinline__ v8f wmma_bf16(v16bf a, v16bf b, v8f c) {
  return __builtin_amdgcn_wmma_f32_16x16x32_bf16(false, a, false, b, (short)0, c,
                                                 false, false);
}

// Pre-swizzle a 512(K) x 16(N) weight slab (fp32 row-major, leading dim ldw)
// into WMMA B-operand order: LDS layout [kblk(16)][lane(32)][i(16)] u16.
// Each lane then reads its v16bf as two contiguous 16B ds_load_b128 chunks.
__device__ void fill_B_slab(unsigned short* lds, const float* __restrict__ W,
                            int colBase, int ldw, int tid, int nthreads) {
  for (int idx = tid; idx < 16 * 32 * 16; idx += nthreads) {
    int i    = idx & 15;
    int lane = (idx >> 4) & 31;
    int kblk = idx >> 9;
    int k = kblk * 32 + kOf(i, lane);
    lds[idx] = f2bf(W[k * ldw + colBase + (lane & 15)]);
  }
}

__device__ __forceinline__ v16bf read_B(const unsigned short* lds, int kblk, int lane) {
  const v8u* p = (const v8u*)(lds + (((kblk << 5) + lane) << 4));
  v8u lo = p[0], hi = p[1];
  v16u t;
#pragma unroll
  for (int i = 0; i < 8; ++i) { t[i] = lo[i]; t[i + 8] = hi[i]; }
  return __builtin_bit_cast(v16bf, t);
}

// A tile (16 rows x 32 K) from a bf16 row-major source: two 16B global loads/lane.
__device__ __forceinline__ v16bf load_A_bf16(const unsigned short* __restrict__ src,
                                             int row0, int ld, int k0, int lane) {
  const unsigned short* p = src + (row0 + (lane & 15)) * ld + k0 + ((lane & 16) ? 8 : 0);
  v8u lo = *(const v8u*)p;
  v8u hi = *(const v8u*)(p + 16);
  v16u t;
#pragma unroll
  for (int i = 0; i < 8; ++i) { t[i] = lo[i]; t[i + 8] = hi[i]; }
  return __builtin_bit_cast(v16bf, t);
}

__device__ __forceinline__ float sigmoidf_(float x) { return 1.0f / (1.0f + __expf(-x)); }

// G is kept in WMMA C-fragment order so both producer and consumer move it as
// 32B vectors:  G[t][g][strip(32)][tile(4)][lane(32)][r(8)]  fp32.
__device__ __forceinline__ size_t gIdx(int t, int g, int strip, int tile, int lane) {
  return ((((size_t)(t * 4 + g) * 32 + strip) * 4 + tile) * 32 + lane) * 8;
}

// ---------- phase 0: cast X to bf16 once (bandwidth-bound, ~96 MB total) ----------
__global__ __launch_bounds__(256) void lstm_xcast(const float* __restrict__ X,
                                                  unsigned short* __restrict__ Xbf) {
  const int i = (blockIdx.x * 256 + threadIdx.x) * 8;   // grid covers exactly T*B*I
  v4f a = *(const v4f*)(X + i);
  v4f b = *(const v4f*)(X + i + 4);
  v8u o;
#pragma unroll
  for (int j = 0; j < 4; ++j) { o[j] = f2bf(a[j]); o[j + 4] = f2bf(b[j]); }
  *(v8u*)(Xbf + i) = o;
}

// ---------- phase 1: G[t][g] = x @ Wx_g + b_g  (throughput GEMM) ----------
// Block = (16-col strip s covering ALL 4 gates, chunk of 256 X-rows).
// 64 KB LDS of pre-swizzled Wx; each wave does 4 row tiles, 4 WMMAs per A-load.
__global__ __launch_bounds__(128) void lstm_phase1(
    const unsigned short* __restrict__ Xbf,
    const float* __restrict__ Wxi, const float* __restrict__ Wxf,
    const float* __restrict__ Wxo, const float* __restrict__ Wxc,
    const float* __restrict__ bi,  const float* __restrict__ bf_,
    const float* __restrict__ bo,  const float* __restrict__ bc,
    float* __restrict__ G) {
  __shared__ unsigned short ldsB[4 * 16 * 32 * 16];   // 64 KB (4 gates)
  const int tid = threadIdx.x, lane = tid & 31, wave = tid >> 5;
  const int s = blockIdx.x & 31;
  const int chunk = blockIdx.x >> 5;
  const int colBase = s * 16;
  fill_B_slab(ldsB + 0 * 8192, Wxi, colBase, HDIM, tid, 128);
  fill_B_slab(ldsB + 1 * 8192, Wxf, colBase, HDIM, tid, 128);
  fill_B_slab(ldsB + 2 * 8192, Wxo, colBase, HDIM, tid, 128);
  fill_B_slab(ldsB + 3 * 8192, Wxc, colBase, HDIM, tid, 128);
  __syncthreads();
  const int col = colBase + (lane & 15);
  float bv[4] = {bi[col], bf_[col], bo[col], bc[col]};
  for (int q = 0; q < 4; ++q) {
    const int row0 = chunk * 256 + (wave * 4 + q) * 16;
    v8f acc[4];
#pragma unroll
    for (int g = 0; g < 4; ++g)
#pragma unroll
      for (int r = 0; r < 8; ++r) acc[g][r] = bv[g];   // fold bias into C
#pragma unroll 4
    for (int kb = 0; kb < 16; ++kb) {
      v16bf a = load_A_bf16(Xbf, row0, IDIM, kb * 32, lane);  // shared by 4 gates
#pragma unroll
      for (int g = 0; g < 4; ++g) {
        v16bf b = read_B(ldsB + g * 8192, kb, lane);
        acc[g] = wmma_bf16(a, b, acc[g]);
      }
    }
    const int t    = row0 >> 6;          // 16-row tiles never straddle a batch block
    const int tile = (row0 >> 4) & 3;
#pragma unroll
    for (int g = 0; g < 4; ++g)
      *(v8f*)(G + gIdx(t, g, s, tile, lane)) = acc[g];   // 2x global_store_b128
  }
}

// ---------- grid-wide barrier for the persistent scan ----------
__device__ __forceinline__ void grid_barrier(unsigned* counter, unsigned* gen,
                                             unsigned nblocks) {
  __syncthreads();
  if (threadIdx.x == 0) {
    __threadfence();                                   // release h writes
    volatile unsigned* vgen = (volatile unsigned*)gen;
    unsigned gseen = *vgen;                            // read gen BEFORE arriving
    if (atomicAdd(counter, 1u) == nblocks - 1) {
      *(volatile unsigned*)counter = 0u;
      __threadfence();
      atomicAdd(gen, 1u);
    } else {
      while (*vgen == gseen) __builtin_amdgcn_s_sleep(2);
    }
    __threadfence();                                   // acquire
  }
  __syncthreads();
}

// ---------- phase 2: persistent sequential scan ----------
// 32 blocks (one per 16-wide hidden column strip, covering all 4 gates),
// 4 waves per block (one per 16-row batch tile). Wh slabs live in LDS in
// WMMA order; h ping-pongs through two bf16 global buffers; c stays in VGPRs.
__global__ __launch_bounds__(128) void lstm_phase2(
    const float* __restrict__ Whi, const float* __restrict__ Whf,
    const float* __restrict__ Who, const float* __restrict__ Whc,
    const float* __restrict__ G,
    float* __restrict__ out,               // [T][B][H] ++ Hf[B][H] ++ Cf[B][H]
    unsigned short* __restrict__ hbuf,     // 2 * B*H bf16 (ping-pong)
    unsigned* __restrict__ bar) {
  __shared__ unsigned short ldsB[4 * 16 * 32 * 16];   // 64 KB (4 gates)
  const int tid = threadIdx.x, lane = tid & 31, wave = tid >> 5;
  const int sIdx = blockIdx.x;
  const int colBase = sIdx * 16;
  fill_B_slab(ldsB + 0 * 8192, Whi, colBase, HDIM, tid, 128);
  fill_B_slab(ldsB + 1 * 8192, Whf, colBase, HDIM, tid, 128);
  fill_B_slab(ldsB + 2 * 8192, Who, colBase, HDIM, tid, 128);
  fill_B_slab(ldsB + 3 * 8192, Whc, colBase, HDIM, tid, 128);
  const int row0 = wave * 16;
  const int col  = colBase + (lane & 15);
  const int mOfs = (lane & 16) ? 8 : 0;
  float creg[8], hreg[8];                  // cell state lives in registers
#pragma unroll
  for (int r = 0; r < 8; ++r) {            // h0 = 0 (owner-exclusive), c0 = 0
    hbuf[(row0 + r + mOfs) * HDIM + col] = 0;
    creg[r] = 0.0f;
    hreg[r] = 0.0f;
  }
  grid_barrier(bar, bar + 1, gridDim.x);

  for (int t = 0; t < TSTEPS; ++t) {
    const unsigned short* hread = hbuf + (t & 1) * (BATCH * HDIM);
    unsigned short* hwrite      = hbuf + ((t + 1) & 1) * (BATCH * HDIM);
    v8f acc[4];
#pragma unroll
    for (int g = 0; g < 4; ++g)            // preactivations: one v8f load each
      acc[g] = *(const v8f*)(G + gIdx(t, g, sIdx, wave, lane));
#pragma unroll 4
    for (int kb = 0; kb < 16; ++kb) {
      v16bf a = load_A_bf16(hread, row0, HDIM, kb * 32, lane);  // shared by 4 gates
#pragma unroll
      for (int g = 0; g < 4; ++g) {
        v16bf b = read_B(ldsB + g * 8192, kb, lane);
        acc[g] = wmma_bf16(a, b, acc[g]);
      }
    }
    float* outT = out + (size_t)t * BATCH * HDIM;
#pragma unroll
    for (int r = 0; r < 8; ++r) {
      const int off = (row0 + r + mOfs) * HDIM + col;
      const float iv = sigmoidf_(acc[0][r]);
      const float fv = sigmoidf_(acc[1][r]);
      const float ov = sigmoidf_(acc[2][r]);
      const float ct = tanhf(acc[3][r]);
      const float cn = fv * creg[r] + iv * ct;
      const float hn = ov * tanhf(cn);
      creg[r] = cn;
      hreg[r] = hn;
      outT[off]   = hn;
      hwrite[off] = f2bf(hn);
    }
    grid_barrier(bar, bar + 1, gridDim.x);   // publish h for step t+1
  }
#pragma unroll
  for (int r = 0; r < 8; ++r) {              // Hf / Cf, once, after the scan
    const int off = (row0 + r + mOfs) * HDIM + col;
    out[(size_t)TSTEPS * BATCH * HDIM + off]       = hreg[r];
    out[(size_t)(TSTEPS + 1) * BATCH * HDIM + off] = creg[r];
  }
}

// ---------- launch ----------
extern "C" void kernel_launch(void* const* d_in, const int* in_sizes, int n_in,
                              void* d_out, int out_size, void* d_ws, size_t ws_size,
                              hipStream_t stream) {
  const float* X   = (const float*)d_in[0];
  const float* Wxi = (const float*)d_in[1];
  const float* Whi = (const float*)d_in[2];
  const float* bi  = (const float*)d_in[3];
  const float* Wxf = (const float*)d_in[4];
  const float* Whf = (const float*)d_in[5];
  const float* bf_ = (const float*)d_in[6];
  const float* Wxo = (const float*)d_in[7];
  const float* Who = (const float*)d_in[8];
  const float* bo  = (const float*)d_in[9];
  const float* Wxc = (const float*)d_in[10];
  const float* Whc = (const float*)d_in[11];
  const float* bc  = (const float*)d_in[12];
  float* out = (float*)d_out;

  // workspace layout
  char* ws = (char*)d_ws;
  float* G = (float*)ws;                                   // T*4*B*H fp32 = 256 MB
  size_t off = (size_t)TSTEPS * 4 * BATCH * HDIM * sizeof(float);
  unsigned short* Xbf = (unsigned short*)(ws + off);       // T*B*I bf16 = 32 MB
  off += (size_t)TSTEPS * BATCH * IDIM * sizeof(unsigned short);
  unsigned short* hbuf = (unsigned short*)(ws + off);      // 2*B*H bf16
  off += (size_t)2 * BATCH * HDIM * sizeof(unsigned short);
  unsigned* bar = (unsigned*)(ws + off);                   // counter + generation

  hipMemsetAsync(bar, 0, 16, stream);

  // phase 0: cast X to bf16 (T*B*I / (256*8) blocks)
  lstm_xcast<<<(TSTEPS * BATCH * IDIM) / (256 * 8), 256, 0, stream>>>(X, Xbf);
  // phase 1: 32 col strips (x 4 gates in-block) * (32768/256) row chunks
  lstm_phase1<<<32 * (TSTEPS * BATCH / 256), 128, 0, stream>>>(
      Xbf, Wxi, Wxf, Wxo, Wxc, bi, bf_, bo, bc, G);
  // phase 2: persistent scan, 32 blocks cover the 2048-wide gate output
  lstm_phase2<<<32, 128, 0, stream>>>(Whi, Whf, Who, Whc, G, out,
                                      hbuf, bar);
}
